// MultiHeadAttention_56959856280444
// MI455X (gfx1250) — matrix operand order repaired and checked by reading, not compiled, not run on hardware
//
#include <hip/hip_runtime.h>
#include <stdint.h>

// MI455X / gfx1250 multi-head causal attention (B=2,S=4096,D=768,H=12,dk=64)
// bf16 WMMA (v_wmma_f32_16x16x32_bf16) + fp32 accumulation.
//  - Projection GEMMs: double-buffered W tile staged via GLOBAL_LOAD_ASYNC_TO_LDS_B128
//    (ASYNCcnt), straightline steady-state loop, load/WMMA decoupled so the chunk
//    needs a single dscnt/loadcnt wait instead of one per WMMA.
//  - V projection stores V^T [b,h,dk,S] so attention P@V B-fragments are contiguous
//    global loads (L2-resident); attention kernel is barrier-free.
// Workspace: ~54 MB.

typedef __bf16 bf16;
typedef __attribute__((ext_vector_type(16))) __bf16 v16bf;
typedef __attribute__((ext_vector_type(8)))  __bf16 v8bf;
typedef __attribute__((ext_vector_type(8)))  float  v8f;

#define B_  2
#define S_  4096
#define D_  768
#define H_  12
#define DK_ 64

#define NEG_INF (-__builtin_inff())

__global__ void mha_cvt_bf16(const float* __restrict__ in, bf16* __restrict__ out, int n) {
  int i = blockIdx.x * blockDim.x + threadIdx.x;
  if (i < n) out[i] = (bf16)in[i];
}

__device__ __forceinline__ v16bf ld_frag_pair(const bf16* p0, const bf16* p1) {
  v8bf lo = *(const v8bf*)p0;
  v8bf hi = *(const v8bf*)p1;
  v16bf f;
#pragma unroll
  for (int t = 0; t < 8; ++t) { f[t] = lo[t]; f[8 + t] = hi[t]; }
  return f;
}

// Compute one 64-wide K chunk: 2 A-frags + 8 B-frags loaded up front, then 8 WMMAs.
__device__ __forceinline__ void gemm_chunk(const bf16* __restrict__ arow, int kc,
                                           const bf16 (*__restrict__ Wsb)[72],
                                           int half, int l16, v8f acc[4]) {
  v16bf a[2];
#pragma unroll
  for (int k2 = 0; k2 < 2; ++k2)
    a[k2] = ld_frag_pair(arow + kc + k2 * 32 + half * 8,
                         arow + kc + k2 * 32 + half * 8 + 16);
  v16bf bfr[2][4];
#pragma unroll
  for (int k2 = 0; k2 < 2; ++k2)
#pragma unroll
    for (int sub = 0; sub < 4; ++sub) {
      const bf16* wl = &Wsb[sub * 16 + l16][k2 * 32 + half * 8];
      bfr[k2][sub] = ld_frag_pair(wl, wl + 16);
    }
#pragma unroll
  for (int k2 = 0; k2 < 2; ++k2)
#pragma unroll
    for (int sub = 0; sub < 4; ++sub)
      acc[sub] = __builtin_amdgcn_wmma_f32_16x16x32_bf16(
          false, a[k2], false, bfr[k2][sub], (short)0, acc[sub], false, false);
}

// Out = scale * (A @ W^T).  A:[M,K]=[8192,768] row-major, W:[N,K]=[768,768] row-major.
// Block: 256 threads (8 waves), tile 128(M) x 64(N), K-chunk 64, W tile double-buffered
// in LDS via async global->LDS DMA. vtrans=1 stores output per-head transposed (for V).
__global__ __launch_bounds__(256)
void mha_gemm_xwT(const bf16* __restrict__ A, const bf16* __restrict__ W,
                  bf16* __restrict__ Out, float scale, int vtrans) {
  __shared__ bf16 Ws[2][64][72];   // [buf][w-col][K-elem], padded rows (144B)

  const int lane = threadIdx.x & 31;
  const int wave = threadIdx.x >> 5;
  const int half = lane >> 4;
  const int l16  = lane & 15;
  const int row0 = blockIdx.y * 128 + wave * 16;
  const int col0 = blockIdx.x * 64;
  const int K = D_;

  // cooperative async staging: thread -> (W row, 32B segment)
  const int srow = threadIdx.x >> 2;
  const int sseg = (threadIdx.x & 3) * 16;
  const bf16* wsrc = W + (size_t)(col0 + srow) * K + sseg;

  { // issue chunk 0 into buffer 0
    unsigned l = (unsigned)(uintptr_t)&Ws[0][srow][sseg];
    unsigned long long g = (unsigned long long)(uintptr_t)wsrc;
    asm volatile("global_load_async_to_lds_b128 %0, %1, off\n\t"
                 "global_load_async_to_lds_b128 %0, %1, off offset:16"
                 :: "v"(l), "v"(g) : "memory");
  }

  v8f acc[4] = {};
  const bf16* arow = A + (size_t)(row0 + l16) * K;

  int kc = 0;
  for (; kc < K - 64; kc += 64) {           // steady state: always prefetch next
    const int cur = (kc >> 6) & 1;
    __syncthreads();                        // all waves done reading buffer cur^1
    {
      unsigned l = (unsigned)(uintptr_t)&Ws[cur ^ 1][srow][sseg];
      unsigned long long g = (unsigned long long)(uintptr_t)(wsrc + kc + 64);
      asm volatile("global_load_async_to_lds_b128 %0, %1, off\n\t"
                   "global_load_async_to_lds_b128 %0, %1, off offset:16"
                   :: "v"(l), "v"(g) : "memory");
    }
    asm volatile("s_wait_asynccnt 0x2" ::: "memory");   // chunk kc's 2 ops retired
    __syncthreads();                        // buffer cur fully resident
    gemm_chunk(arow, kc, Ws[cur], half, l16, acc);
  }
  { // last chunk (no prefetch)
    const int cur = (kc >> 6) & 1;
    __syncthreads();
    asm volatile("s_wait_asynccnt 0x0" ::: "memory");
    __syncthreads();
    gemm_chunk(arow, kc, Ws[cur], half, l16, acc);
  }

  if (vtrans) {                             // V^T: [b][h][dd][s]
#pragma unroll
    for (int sub = 0; sub < 4; ++sub) {
      int c = col0 + sub * 16 + l16;
      int h = c >> 6, dd = c & (DK_ - 1);
#pragma unroll
      for (int r = 0; r < 8; ++r) {
        int m = row0 + r + 8 * half;
        int b = m >> 12, s = m & (S_ - 1);
        Out[((size_t)(b * H_ + h) * DK_ + dd) * S_ + s] = (bf16)(acc[sub][r] * scale);
      }
    }
  } else {
#pragma unroll
    for (int sub = 0; sub < 4; ++sub) {
      int c = col0 + sub * 16 + l16;
#pragma unroll
      for (int r = 0; r < 8; ++r) {
        int m = row0 + r + 8 * half;
        Out[(size_t)m * D_ + c] = (bf16)(acc[sub][r] * scale);
      }
    }
  }
}

// Flash attention, causal. Grid: (S/128, H, B). Block: 256 threads = 8 waves.
// Waves fully independent (per-wave causal bound, per-wave LDS P relay, no barriers).
__global__ __launch_bounds__(256)
void mha_flash_attn(const bf16* __restrict__ Q,    // [B,S,D], pre-scaled by 1/sqrt(dk)
                    const bf16* __restrict__ K,    // [B,S,D]
                    const bf16* __restrict__ Vt,   // [B,H,DK,S]
                    float* __restrict__ O) {       // [B,S,D]
  __shared__ bf16 P_lds[8][16][72];   // per-wave D-layout -> A-layout relay

  const int lane = threadIdx.x & 31;
  const int wave = threadIdx.x >> 5;
  const int half = lane >> 4;
  const int l16  = lane & 15;
  const int h    = blockIdx.y;
  const int b    = blockIdx.z;
  const int qrow = blockIdx.x * 128 + wave * 16;

  const size_t base = (size_t)b * S_ * D_ + (size_t)h * DK_;
  const bf16* Vp = Vt + (size_t)(b * H_ + h) * DK_ * S_;

  // Q A-fragments, 2 chunks covering d_k = 64
  v16bf aQ[2];
  {
    const bf16* qp = Q + base + (size_t)(qrow + l16) * D_;
#pragma unroll
    for (int c = 0; c < 2; ++c)
      aQ[c] = ld_frag_pair(qp + c * 32 + half * 16, qp + c * 32 + half * 16 + 8);
  }

  float mrow[8], lrow[8];
  v8f oacc[4] = {};
#pragma unroll
  for (int r = 0; r < 8; ++r) { mrow[r] = NEG_INF; lrow[r] = 0.f; }

  const int jend = qrow + 16;   // this wave's causal bound
  for (int j = 0; j < jend; j += 64) {
    if (j + 64 < jend)   // global_prefetch_b8 of next K block
      __builtin_prefetch(K + base + (size_t)(j + 64) * D_, 0, 1);

    // Scores: S = Q @ K^T (16 rows x 64 keys), 1/sqrt(dk) folded into Q
    v8f s[4] = {};
#pragma unroll
    for (int sub = 0; sub < 4; ++sub) {
      const bf16* kp = K + base + (size_t)(j + sub * 16 + l16) * D_ + half * 16;
#pragma unroll
      for (int c = 0; c < 2; ++c) {
        v16bf bfr = *(const v16bf*)(kp + c * 32);
        s[sub] = __builtin_amdgcn_wmma_f32_16x16x32_bf16(
            false, aQ[c], false, bfr, (short)0, s[sub], false, false);
      }
    }

    if (j + 64 > qrow) {   // causal mask: only the diagonal block (uniform branch)
#pragma unroll
      for (int sub = 0; sub < 4; ++sub) {
        int kn = j + sub * 16 + l16;
#pragma unroll
        for (int r = 0; r < 8; ++r) {
          int qm = qrow + r + 8 * half;
          if (kn > qm) s[sub][r] = NEG_INF;
        }
      }
    }

    // Online softmax; key dim lives across 16-lane halves -> shfl_xor reductions
#pragma unroll
    for (int r = 0; r < 8; ++r) {
      float v = fmaxf(fmaxf(s[0][r], s[1][r]), fmaxf(s[2][r], s[3][r]));
#pragma unroll
      for (int m = 1; m < 16; m <<= 1) v = fmaxf(v, __shfl_xor(v, m, 32));
      float mnew  = fmaxf(mrow[r], v);
      float alpha = __expf(mrow[r] - mnew);
      mrow[r] = mnew;
      float sum = 0.f;
#pragma unroll
      for (int sub = 0; sub < 4; ++sub) {
        float p = __expf(s[sub][r] - mnew);
        s[sub][r] = p;
        sum += p;
      }
#pragma unroll
      for (int m = 1; m < 16; m <<= 1) sum += __shfl_xor(sum, m, 32);
      lrow[r] = lrow[r] * alpha + sum;
#pragma unroll
      for (int sub = 0; sub < 4; ++sub) oacc[sub][r] *= alpha;
    }

    // P: D-layout -> per-wave LDS -> A-layout (LDS ops in-order within a wave)
#pragma unroll
    for (int sub = 0; sub < 4; ++sub)
#pragma unroll
      for (int r = 0; r < 8; ++r)
        P_lds[wave][r + 8 * half][sub * 16 + l16] = (bf16)s[sub][r];

    // Both aP fragments first (one dscnt wait), then 8 WMMAs with pipelined V^T loads.
    v16bf aP[2];
#pragma unroll
    for (int c = 0; c < 2; ++c) {
      const bf16* pp = &P_lds[wave][l16][c * 32 + half * 16];
      aP[c] = ld_frag_pair(pp, pp + 8);
    }
#pragma unroll
    for (int c = 0; c < 2; ++c)
#pragma unroll
      for (int sub = 0; sub < 4; ++sub) {
        const bf16* vp = Vp + (size_t)(sub * 16 + l16) * S_ + j + c * 32 + half * 16;
        v16bf bV = *(const v16bf*)vp;
        oacc[sub] = __builtin_amdgcn_wmma_f32_16x16x32_bf16(
            false, aP[c], false, bV, (short)0, oacc[sub], false, false);
      }
  }

  // Normalize and store fp32
#pragma unroll
  for (int sub = 0; sub < 4; ++sub) {
    int d = sub * 16 + l16;
#pragma unroll
    for (int r = 0; r < 8; ++r) {
      int m = qrow + r + 8 * half;
      O[base + (size_t)m * D_ + d] = oacc[sub][r] / lrow[r];
    }
  }
}

extern "C" void kernel_launch(void* const* d_in, const int* in_sizes, int n_in,
                              void* d_out, int out_size, void* d_ws, size_t ws_size,
                              hipStream_t stream) {
  (void)in_sizes; (void)n_in; (void)out_size; (void)ws_size;
  const float* x  = (const float*)d_in[0];
  const float* Wq = (const float*)d_in[1];
  const float* Wk = (const float*)d_in[2];
  const float* Wv = (const float*)d_in[3];
  // d_in[4] = num_heads (compile-time constant H_=12 here)
  float* out = (float*)d_out;

  const size_t MB = (size_t)B_ * S_ * D_;   // 6,291,456 elements
  bf16* Xb  = (bf16*)d_ws;
  bf16* Qb  = Xb + MB;
  bf16* Kb  = Qb + MB;
  bf16* Vtb = Kb + MB;                      // [B,H,DK,S]
  bf16* Wqb = Vtb + MB;
  bf16* Wkb = Wqb + (size_t)D_ * D_;
  bf16* Wvb = Wkb + (size_t)D_ * D_;
  // total ws: 4*MB*2 + 3*D*D*2 ~= 54 MB

  {
    int n = (int)MB;
    mha_cvt_bf16<<<(n + 255) / 256, 256, 0, stream>>>(x, Xb, n);
    int nw = D_ * D_;
    mha_cvt_bf16<<<(nw + 255) / 256, 256, 0, stream>>>(Wq, Wqb, nw);
    mha_cvt_bf16<<<(nw + 255) / 256, 256, 0, stream>>>(Wk, Wkb, nw);
    mha_cvt_bf16<<<(nw + 255) / 256, 256, 0, stream>>>(Wv, Wvb, nw);
  }

  dim3 gg(D_ / 64, (B_ * S_) / 128);
  const float qscale = 0.125f;  // 1/sqrt(dk), folded into Q
  mha_gemm_xwT<<<gg, 256, 0, stream>>>(Xb, Wqb, Qb, qscale, 0);
  mha_gemm_xwT<<<gg, 256, 0, stream>>>(Xb, Wkb, Kb, 1.0f, 0);
  mha_gemm_xwT<<<gg, 256, 0, stream>>>(Xb, Wvb, Vtb, 1.0f, 1);

  dim3 ga(S_ / 128, H_, B_);
  mha_flash_attn<<<ga, 256, 0, stream>>>(Qb, Kb, Vtb, out);
}
